// ChannelGroupedCrossAttention_65592740545005
// MI455X (gfx1250) — matrix operand order repaired and checked by reading, not compiled, not run on hardware
//
#include <hip/hip_runtime.h>

#define CC   19
#define NPTS 361
#define BB   32
#define DD   256
#define HH   8
#define HDIM 32
#define NPAD 384   // 12 * 32 key padding
#define QPAD 32    // query padding

typedef __attribute__((ext_vector_type(16))) __bf16 v16bf;
typedef __attribute__((ext_vector_type(8)))  __bf16 v8bf;
typedef __attribute__((ext_vector_type(8)))  float  v8f;

static __device__ __forceinline__ unsigned short f2bf(float f) {
  unsigned int u = __float_as_uint(f);
  u += 0x7FFFu + ((u >> 16) & 1u);
  return (unsigned short)(u >> 16);
}
static __device__ __forceinline__ float bf2f(unsigned short h) {
  return __uint_as_float(((unsigned int)h) << 16);
}
static __device__ __forceinline__ v8f zero8f() {
  v8f z;
#pragma unroll
  for (int i = 0; i < 8; ++i) z[i] = 0.f;
  return z;
}
static __device__ __forceinline__ v16bf zero16bf() {
  v16bf z;
#pragma unroll
  for (int i = 0; i < 16; ++i) z[i] = (__bf16)0.f;
  return z;
}
// B-fragment: 32 contiguous bytes per lane (pre-packed / transposed layouts)
static __device__ __forceinline__ v16bf ld_bfrag(const unsigned short* p) {
  return *(const v16bf*)p;
}
// A-fragment: two 16-byte halves (K 16h..16h+7 and K 16h+8..16h+15 for this lane)
static __device__ __forceinline__ v16bf ld_afrag(const unsigned short* p0, const unsigned short* p1) {
  v8bf a = *(const v8bf*)p0;
  v8bf b = *(const v8bf*)p1;
  return __builtin_shufflevector(a, b, 0,1,2,3,4,5,6,7,8,9,10,11,12,13,14,15);
}
static __device__ __forceinline__ v8f wmma_bf16(v16bf a, v16bf b, v8f c) {
  return __builtin_amdgcn_wmma_f32_16x16x32_bf16(false, a, false, b, (short)0, c, false, false);
}

// ---------------- utility kernels ----------------

__global__ void k_cvt_x(const float* __restrict__ x, unsigned short* __restrict__ xb, long long n8) {
  long long i = (long long)blockIdx.x * blockDim.x + threadIdx.x;
  long long stride = (long long)gridDim.x * blockDim.x;
  for (; i < n8; i += stride) {
    const float4* s = (const float4*)(x + i * 8);
    float4 a = s[0], b = s[1];
    uint4 o;
    o.x = (unsigned)f2bf(a.x) | ((unsigned)f2bf(a.y) << 16);
    o.y = (unsigned)f2bf(a.z) | ((unsigned)f2bf(a.w) << 16);
    o.z = (unsigned)f2bf(b.x) | ((unsigned)f2bf(b.y) << 16);
    o.w = (unsigned)f2bf(b.z) | ((unsigned)f2bf(b.w) << 16);
    *(uint4*)(xb + i * 8) = o;
  }
}

// Pack w (row-major [E=256][D=256] per channel) into WMMA B-fragment order:
// dst[((ch*8+kk)*16+nt)*512 + lane*16 + e] = w[ch][nt*16+(lane&15)][kk*32+((lane>>4)<<4)+e]
__global__ void k_pack_w(const float* __restrict__ src, long long chan_stride, int row_off,
                         unsigned short* __restrict__ dst, int nchan) {
  long long total = (long long)nchan * 65536;
  long long i = (long long)blockIdx.x * blockDim.x + threadIdx.x;
  long long stride = (long long)gridDim.x * blockDim.x;
  for (; i < total; i += stride) {
    int ch   = (int)(i >> 16);
    int r    = (int)(i & 65535);
    int kk   = r >> 13;
    int nt   = (r >> 9) & 15;
    int lane = (r >> 4) & 31;
    int e    = r & 15;
    int srcE = nt * 16 + (lane & 15);
    int srcD = kk * 32 + ((lane >> 4) << 4) + e;
    float v = src[(long long)ch * chan_stride + (long long)(row_off + srcE) * DD + srcD];
    dst[i] = f2bf(v);
  }
}

// ---------------- K/V projection: K[c,b] = x[b] @ wk[c]^T, V likewise ----------------
// Block covers 32 tokens; each wave computes a 32x64 tile (2 M-tiles x 4 N-tiles),
// so every B-fragment load feeds two WMMAs. K stored row-major [token][d];
// V stored transposed [d][token]; all NPAD tokens covered (pad rows = bias, finite).
__global__ __launch_bounds__(256)
void k_kv_proj(const unsigned short* __restrict__ xb,
               const unsigned short* __restrict__ wkP,
               const unsigned short* __restrict__ wvP,
               const float* __restrict__ b_in,
               unsigned short* __restrict__ Kbuf,
               unsigned short* __restrict__ Vt) {
  __shared__ __align__(16) unsigned short lA[32 * 264];   // 16,896 B
  __shared__ __align__(16) unsigned short lK[32 * 256];   // 16,384 B
  __shared__ __align__(16) unsigned short lV[256 * 32];   // 16,384 B
  int cb = blockIdx.x;
  int c = cb / BB, b = cb % BB;
  int m0 = blockIdx.y * 32;
  int tid = threadIdx.x;
  // stage 32x256 x-tile into LDS (row stride 264 to avoid bank conflicts)
#pragma unroll
  for (int u = tid; u < 512; u += 256) {
    int m = u >> 4, chunk = u & 15;
    int token = m0 + m;
    uint4 d0 = make_uint4(0u,0u,0u,0u), d1 = d0;
    if (token < NPTS) {
      const uint4* s = (const uint4*)(xb + ((long long)b * NPTS + token) * DD + chunk * 16);
      d0 = s[0]; d1 = s[1];
    }
    uint4* d = (uint4*)(lA + m * 264 + chunk * 16);
    d[0] = d0; d[1] = d1;
  }
  __syncthreads();

  int wave = tid >> 5, lane = tid & 31;
  int h4 = lane >> 4, ln = lane & 15;
  bool isV = wave >= 4;
  int nbase = (wave & 3) * 64;
  const unsigned short* wp = isV ? wvP : wkP;
  const unsigned short* wbase = wp + (((long long)c * 8) * 16 + (nbase >> 4)) * 512 + lane * 16;
  v8f acc[2][4];
#pragma unroll
  for (int mt = 0; mt < 2; ++mt)
#pragma unroll
    for (int t = 0; t < 4; ++t) acc[mt][t] = zero8f();

#pragma unroll
  for (int kk = 0; kk < 8; ++kk) {
    v16bf af[2];
#pragma unroll
    for (int mt = 0; mt < 2; ++mt) {
      const unsigned short* a0 = lA + (mt * 16 + ln) * 264 + kk * 32 + h4 * 16;
      af[mt] = ld_afrag(a0, a0 + 8);
    }
    v16bf bf[4];
    const unsigned short* bp = wbase + (long long)kk * 16 * 512;
    if (kk < 7) __builtin_prefetch(bp + 16 * 512, 0, 1);
#pragma unroll
    for (int t = 0; t < 4; ++t) bf[t] = ld_bfrag(bp + (long long)t * 512);
#pragma unroll
    for (int t = 0; t < 4; ++t)
#pragma unroll
      for (int mt = 0; mt < 2; ++mt) acc[mt][t] = wmma_bf16(af[mt], bf[t], acc[mt][t]);
  }
  // bias + transpose through LDS
  const float* bias = b_in + (long long)c * 768 + (isV ? 512 : 256);
#pragma unroll
  for (int t = 0; t < 4; ++t) {
    int col = nbase + t * 16 + ln;
    float bv = bias[col];
#pragma unroll
    for (int mt = 0; mt < 2; ++mt) {
      if (!isV) {
#pragma unroll
        for (int r = 0; r < 8; ++r)
          lK[(mt * 16 + h4 * 8 + r) * 256 + col] = f2bf(acc[mt][t][r] + bv);
      } else {
        unsigned short tmp[8];
#pragma unroll
        for (int r = 0; r < 8; ++r) tmp[r] = f2bf(acc[mt][t][r] + bv);
        uint4 pk;
        pk.x = (unsigned)tmp[0] | ((unsigned)tmp[1] << 16);
        pk.y = (unsigned)tmp[2] | ((unsigned)tmp[3] << 16);
        pk.z = (unsigned)tmp[4] | ((unsigned)tmp[5] << 16);
        pk.w = (unsigned)tmp[6] | ((unsigned)tmp[7] << 16);
        *(uint4*)(lV + col * 32 + mt * 16 + h4 * 8) = pk;
      }
    }
  }
  __syncthreads();
  // cooperative coalesced stores (unconditional: NPAD rows are allocated)
#pragma unroll
  for (int u = tid; u < 512; u += 256) {
    int m = u >> 4, chunk = u & 15;
    int token = m0 + m;
    uint4* d = (uint4*)(Kbuf + (((long long)c * BB + b) * NPAD + token) * DD + chunk * 16);
    const uint4* s = (const uint4*)(lK + m * 256 + chunk * 16);
    d[0] = s[0]; d[1] = s[1];
  }
  {
    int col = tid;
    long long base = (((long long)c * BB + b) * DD + col) * NPAD + m0;
    uint4* d = (uint4*)(Vt + base);
    const uint4* s = (const uint4*)(lV + col * 32);
    d[0] = s[0]; d[1] = s[1]; d[2] = s[2]; d[3] = s[3];
  }
}

// ---------------- Q projection: Q[c,b,j] = x[b, c*19+j] @ wq[c]^T ----------------
__global__ __launch_bounds__(256)
void k_q_proj(const unsigned short* __restrict__ xb,
              const unsigned short* __restrict__ wqP,
              const float* __restrict__ b_in,
              unsigned short* __restrict__ Qbuf) {
  int cb = blockIdx.x;
  int c = cb / BB, b = cb % BB;
  int tid = threadIdx.x;
  int wave = tid >> 5, lane = tid & 31;
  int h4 = lane >> 4, ln = lane & 15;
  int mt = wave >> 2;
  int nbase = (wave & 3) * 64;
  int qa = mt * 16 + ln;  // A-fragment row
  v8f acc[4];
#pragma unroll
  for (int t = 0; t < 4; ++t) acc[t] = zero8f();

#pragma unroll
  for (int kk = 0; kk < 8; ++kk) {
    v16bf bf[4];
    long long bbase = (((long long)c * 8 + kk) * 16 + (nbase >> 4)) * 512 + lane * 16;
#pragma unroll
    for (int t = 0; t < 4; ++t) bf[t] = ld_bfrag(wqP + bbase + (long long)t * 512);
    v16bf af;
    if (qa < CC) {
      const unsigned short* a0 = xb + ((long long)b * NPTS + c * CC + qa) * DD + kk * 32 + h4 * 16;
      af = ld_afrag(a0, a0 + 8);
    } else {
      af = zero16bf();
    }
#pragma unroll
    for (int t = 0; t < 4; ++t) acc[t] = wmma_bf16(af, bf[t], acc[t]);
  }
#pragma unroll
  for (int t = 0; t < 4; ++t) {
    int col = nbase + t * 16 + ln;
    float bv = b_in[(long long)c * 768 + col];
#pragma unroll
    for (int r = 0; r < 8; ++r) {
      int qq = mt * 16 + h4 * 8 + r;
      if (qq < CC)
        Qbuf[(((long long)c * BB + b) * QPAD + qq) * DD + col] = f2bf(acc[t][r] + bv);
    }
  }
}

// ---------------- attention: per (c,b), 2 heads per block (one per wave) ----------------
__global__ __launch_bounds__(64)
void k_attn(const unsigned short* __restrict__ Qbuf,
            const unsigned short* __restrict__ Kbuf,
            const unsigned short* __restrict__ Vt,
            unsigned short* __restrict__ ctx) {
  __shared__ __align__(16) unsigned short sAttn[2 * QPAD * NPAD];  // 48 KB
  int cb = blockIdx.x;
  int c = cb / BB, b = cb % BB;
  int wave = threadIdx.x >> 5, lane = threadIdx.x & 31;
  int h = blockIdx.y * 2 + wave;
  int h4 = lane >> 4, ln = lane & 15;
  unsigned short* aL = sAttn + wave * QPAD * NPAD;
  const unsigned short* Qb = Qbuf + ((long long)c * BB + b) * QPAD * DD + h * HDIM;
  const unsigned short* Kb = Kbuf + ((long long)c * BB + b) * NPAD * DD + h * HDIM;
  const unsigned short* Vb = Vt + (((long long)c * BB + b) * DD + h * HDIM) * NPAD;
  const float NEG = -__builtin_inff();
  const float scale = 0.17677669529663687f;

  // preload Q A-fragments (K-dim = HDIM = 32, single step)
  v16bf aq[2];
#pragma unroll
  for (int mt = 0; mt < 2; ++mt) {
    const unsigned short* p = Qb + (long long)(mt * 16 + ln) * DD + h4 * 16;
    aq[mt] = ld_afrag(p, p + 8);
  }

  float mx[2][8];
#pragma unroll
  for (int mt = 0; mt < 2; ++mt)
#pragma unroll
    for (int r = 0; r < 8; ++r) mx[mt][r] = NEG;

  // P1: scores = Q K^T, scale + mask, stash bf16 in LDS, track row max
  for (int nt = 0; nt < 24; ++nt) {
    int key = nt * 16 + ln;
    v16bf bk = ld_bfrag(Kb + (long long)key * DD + h4 * 16);
    int ki = key / CC;
    int kj = key - ki * CC;
#pragma unroll
    for (int mt = 0; mt < 2; ++mt) {
      v8f sc = wmma_bf16(aq[mt], bk, zero8f());
#pragma unroll
      for (int r = 0; r < 8; ++r) {
        int qq = mt * 16 + h4 * 8 + r;
        bool valid = (key < NPTS) && (qq < CC) &&
                     (c == ki || c == kj || qq == ki || qq == kj);
        float v = valid ? sc[r] * scale : NEG;
        mx[mt][r] = fmaxf(mx[mt][r], v);
        aL[(mt * 16 + h4 * 8 + r) * NPAD + key] = f2bf(v);
      }
    }
  }
  __syncthreads();
  // row max over the 16 lanes sharing this row set
#pragma unroll
  for (int mt = 0; mt < 2; ++mt)
#pragma unroll
    for (int r = 0; r < 8; ++r)
      for (int off = 1; off < 16; off <<= 1)
        mx[mt][r] = fmaxf(mx[mt][r], __shfl_xor(mx[mt][r], off, 16));

  // P2: exp(s - max), row sums, rewrite LDS with unnormalized weights
  float inv[2][8];
#pragma unroll
  for (int mt = 0; mt < 2; ++mt)
#pragma unroll
    for (int r = 0; r < 8; ++r) {
      int row = mt * 16 + h4 * 8 + r;
      float m = mx[mt][r];
      float msafe = (m == NEG) ? 0.f : m;
      float s = 0.f;
      for (int nt = 0; nt < 24; ++nt) {
        int key = nt * 16 + ln;
        float v = bf2f(aL[row * NPAD + key]);
        float e = (v == NEG) ? 0.f : __expf(v - msafe);
        s += e;
        aL[row * NPAD + key] = f2bf(e);
      }
      for (int off = 1; off < 16; off <<= 1) s += __shfl_xor(s, off, 16);
      inv[mt][r] = (s > 0.f) ? 1.f / s : 0.f;
    }
  __syncthreads();

  // P3: out = attn @ V (scale rows by 1/sum at the end)
  v8f o[2][2];
#pragma unroll
  for (int mt = 0; mt < 2; ++mt)
#pragma unroll
    for (int n2 = 0; n2 < 2; ++n2) o[mt][n2] = zero8f();
#pragma unroll
  for (int kk = 0; kk < 12; ++kk) {
    v16bf bv[2], av[2];
#pragma unroll
    for (int n2 = 0; n2 < 2; ++n2)
      bv[n2] = ld_bfrag(Vb + (long long)(n2 * 16 + ln) * NPAD + kk * 32 + h4 * 16);
#pragma unroll
    for (int mt = 0; mt < 2; ++mt) {
      const unsigned short* p = aL + (mt * 16 + ln) * NPAD + kk * 32 + h4 * 16;
      av[mt] = ld_afrag(p, p + 8);
    }
#pragma unroll
    for (int mt = 0; mt < 2; ++mt)
#pragma unroll
      for (int n2 = 0; n2 < 2; ++n2) o[mt][n2] = wmma_bf16(av[mt], bv[n2], o[mt][n2]);
  }
#pragma unroll
  for (int mt = 0; mt < 2; ++mt)
#pragma unroll
    for (int n2 = 0; n2 < 2; ++n2)
#pragma unroll
      for (int r = 0; r < 8; ++r) {
        int qq = mt * 16 + h4 * 8 + r;
        if (qq < CC)
          ctx[(((long long)c * BB + b) * QPAD + qq) * DD + h * HDIM + n2 * 16 + ln] =
              f2bf(o[mt][n2][r] * inv[mt][r]);
      }
}

// ---------------- output projection: pout[b, c*19+j] = ctx[c,b,j] @ wo[c]^T + b_out ----------------
__global__ __launch_bounds__(256)
void k_out_proj(const unsigned short* __restrict__ ctx,
                const unsigned short* __restrict__ woP,
                const float* __restrict__ b_out,
                unsigned short* __restrict__ pout) {
  int cb = blockIdx.x;
  int c = cb / BB, b = cb % BB;
  int tid = threadIdx.x;
  int wave = tid >> 5, lane = tid & 31;
  int h4 = lane >> 4, ln = lane & 15;
  int mt = wave >> 2;
  int nbase = (wave & 3) * 64;
  const unsigned short* Ab = ctx + ((long long)c * BB + b) * QPAD * DD;
  v8f acc[4];
#pragma unroll
  for (int t = 0; t < 4; ++t) acc[t] = zero8f();
#pragma unroll
  for (int kk = 0; kk < 8; ++kk) {
    v16bf bf[4];
    long long bbase = (((long long)c * 8 + kk) * 16 + (nbase >> 4)) * 512 + lane * 16;
#pragma unroll
    for (int t = 0; t < 4; ++t) bf[t] = ld_bfrag(woP + bbase + (long long)t * 512);
    const unsigned short* a0 = Ab + (long long)(mt * 16 + ln) * DD + kk * 32 + h4 * 16;
    v16bf af = ld_afrag(a0, a0 + 8);
#pragma unroll
    for (int t = 0; t < 4; ++t) acc[t] = wmma_bf16(af, bf[t], acc[t]);
  }
#pragma unroll
  for (int t = 0; t < 4; ++t) {
    int col = nbase + t * 16 + ln;
    float bv = b_out[(long long)c * DD + col];
#pragma unroll
    for (int r = 0; r < 8; ++r) {
      int qq = mt * 16 + h4 * 8 + r;
      if (qq < CC)
        pout[((long long)b * NPTS + c * CC + qq) * DD + col] = f2bf(acc[t][r] + bv);
    }
  }
}

// ---------------- fuse: out = pout @ wf^T + b_fuse + x (fp32 out) ----------------
__global__ __launch_bounds__(256)
void k_fuse(const unsigned short* __restrict__ pout,
            const unsigned short* __restrict__ wfP,
            const float* __restrict__ b_fuse,
            const float* __restrict__ x,
            float* __restrict__ out) {
  int gm0 = blockIdx.x * 16;
  int tid = threadIdx.x;
  int wave = tid >> 5, lane = tid & 31;
  int h4 = lane >> 4, ln = lane & 15;
  int nbase = wave * 32;
  v8f acc[2];
#pragma unroll
  for (int t = 0; t < 2; ++t) acc[t] = zero8f();
#pragma unroll
  for (int kk = 0; kk < 8; ++kk) {
    v16bf bf[2];
    long long bbase = ((long long)kk * 16 + (nbase >> 4)) * 512 + lane * 16;
#pragma unroll
    for (int t = 0; t < 2; ++t) bf[t] = ld_bfrag(wfP + bbase + (long long)t * 512);
    const unsigned short* a0 = pout + (long long)(gm0 + ln) * DD + kk * 32 + h4 * 16;
    v16bf af = ld_afrag(a0, a0 + 8);
#pragma unroll
    for (int t = 0; t < 2; ++t) acc[t] = wmma_bf16(af, bf[t], acc[t]);
  }
#pragma unroll
  for (int t = 0; t < 2; ++t) {
    int col = nbase + t * 16 + ln;
    float bv = b_fuse[col];
#pragma unroll
    for (int r = 0; r < 8; ++r) {
      long long row = gm0 + h4 * 8 + r;
      out[row * DD + col] = acc[t][r] + bv + x[row * DD + col];
    }
  }
}

// ---------------- launch ----------------
extern "C" void kernel_launch(void* const* d_in, const int* in_sizes, int n_in,
                              void* d_out, int out_size, void* d_ws, size_t ws_size,
                              hipStream_t stream) {
  (void)in_sizes; (void)n_in; (void)out_size; (void)ws_size;
  const float* x      = (const float*)d_in[0];
  const float* w_in   = (const float*)d_in[1];
  const float* b_in   = (const float*)d_in[2];
  const float* w_out  = (const float*)d_in[3];
  const float* b_out  = (const float*)d_in[4];
  const float* w_fuse = (const float*)d_in[5];
  const float* b_fuse = (const float*)d_in[6];
  float* out = (float*)d_out;

  const size_t szQ   = (size_t)CC * BB * QPAD * DD * 2;   // 9,961,472
  const size_t szK   = (size_t)CC * BB * NPAD * DD * 2;   // 119,537,664
  const size_t szV   = szK;
  const size_t szCtx = szQ;
  const size_t szXb  = (size_t)BB * NPTS * DD * 2;        // 5,914,624
  const size_t szW   = (size_t)CC * 65536 * 2;            // 2,490,368
  const size_t szWf  = (size_t)65536 * 2;                 // 131,072
  const size_t szPo  = szXb;

  unsigned char* base = (unsigned char*)d_ws;
  size_t off = 0;
  unsigned short* Qbuf = (unsigned short*)(base + off); off += szQ;
  unsigned short* Kbuf = (unsigned short*)(base + off); off += szK;
  unsigned short* Vt   = (unsigned short*)(base + off); off += szV;
  unsigned short* ctx  = (unsigned short*)(base + off); off += szCtx;
  unsigned short* xb   = (unsigned short*)(base + off); off += szXb;
  unsigned short* wqP  = (unsigned short*)(base + off); off += szW;
  unsigned short* wkP  = (unsigned short*)(base + off); off += szW;
  unsigned short* wvP  = (unsigned short*)(base + off); off += szW;
  unsigned short* woP  = (unsigned short*)(base + off); off += szW;
  unsigned short* wfP  = (unsigned short*)(base + off); off += szWf;
  unsigned short* pout = (unsigned short*)(base + off); off += szPo;

  // 1. convert x to bf16
  {
    long long n8 = (long long)BB * NPTS * DD / 8;
    k_cvt_x<<<(unsigned)((n8 + 255) / 256), 256, 0, stream>>>(x, xb, n8);
  }
  // 2. pack weights into B-fragment order
  k_pack_w<<<2048, 256, 0, stream>>>(w_in, (long long)768 * DD, 0,       wqP, CC);
  k_pack_w<<<2048, 256, 0, stream>>>(w_in, (long long)768 * DD, DD,      wkP, CC);
  k_pack_w<<<2048, 256, 0, stream>>>(w_in, (long long)768 * DD, 2 * DD,  wvP, CC);
  k_pack_w<<<2048, 256, 0, stream>>>(w_out, (long long)DD * DD, 0,       woP, CC);
  k_pack_w<<<512,  256, 0, stream>>>(w_fuse, 0, 0,                       wfP, 1);
  // 3. K/V projections over all NPAD tokens, 32 tokens per block (dominant GEMM)
  k_kv_proj<<<dim3(CC * BB, NPAD / 32), 256, 0, stream>>>(xb, wkP, wvP, b_in, Kbuf, Vt);
  // 4. Q projection
  k_q_proj<<<CC * BB, 256, 0, stream>>>(xb, wqP, b_in, Qbuf);
  // 5. attention (2 heads per block)
  k_attn<<<dim3(CC * BB, HH / 2), 64, 0, stream>>>(Qbuf, Kbuf, Vt, ctx);
  // 6. output projection
  k_out_proj<<<CC * BB, 256, 0, stream>>>(ctx, woP, b_out, pout);
  // 7. fuse + residual
  k_fuse<<<(BB * NPTS) / 16, 256, 0, stream>>>(pout, wfP, b_fuse, x, out);
}
